// FFB_encoder_30391188586865
// MI455X (gfx1250) — matrix operand order repaired and checked by reading, not compiled, not run on hardware
//
#include <hip/hip_runtime.h>
#include <hip/hip_bf16.h>
#include <math.h>

typedef __attribute__((ext_vector_type(16))) _Float16 v16h;
typedef __attribute__((ext_vector_type(8)))  float    v8f;

#define LVLS    4
#define FEAT    8
#define TSIZE   (1 << 19)
#define TMASK   ((1u << 19) - 1u)
#define HID     64
#define ODIM    128
#define NPTS    524288
#define BLK_PTS 128
#define XSTR    72            // halfs per activation row in LDS (64 + 8 pad)

// d_ws layout, in _Float16 units (pre-packed B fragments)
#define WS_PACK_SZ    (LVLS * 2 * 4 * 32 * 16)   // [L][kchunk][ntile][lane][16]
#define WOUT_PACK_SZ  (LVLS * 2 * 8 * 32 * 16)
#define FFN_PACK_SZ   (LVLS * 1 * 4 * 32 * 16)   // K padded 8 -> 32 with zeros
#define WOUT_PACK_OFF (WS_PACK_SZ)
#define FFN_PACK_OFF  (WS_PACK_SZ + WOUT_PACK_SZ)
#define PACK_TOTAL    (WS_PACK_SZ + WOUT_PACK_SZ + FFN_PACK_SZ)

// ---------------------------------------------------------------------------
// Prep: convert f32 weights to f16, laid out directly in the WMMA B-fragment
// order for V_WMMA_F32_16X16X32_F16:
//   element e of lane's v16h  <->  K = 32*kchunk + 16*(e>>3) + 8*(lane>>4) + (e&7) (Ws/Wout)
//   column                    <->  16*ntile + (lane&15)
// ---------------------------------------------------------------------------
__global__ __launch_bounds__(256) void ffb_prep(const float* __restrict__ ffn,
                                                const float* __restrict__ Ws,
                                                const float* __restrict__ Wout,
                                                _Float16* __restrict__ wpack) {
  int idx = blockIdx.x * blockDim.x + threadIdx.x;
  if (idx < WS_PACK_SZ) {
    int l = idx >> 12;               // / (2*4*32*16)
    int c = (idx >> 11) & 1;
    int n = (idx >> 9) & 3;
    int lane = (idx >> 4) & 31;
    int e = idx & 15;
    int k = 32 * c + ((e & 8) << 1) + ((lane >> 4) << 3) + (e & 7);
    int col = 16 * n + (lane & 15);
    wpack[idx] = (_Float16)Ws[(l * 64 + k) * 64 + col];
  } else if (idx < WS_PACK_SZ + WOUT_PACK_SZ) {
    int i2 = idx - WS_PACK_SZ;
    int l = i2 >> 13;                // / (2*8*32*16)
    int c = (i2 >> 12) & 1;
    int n = (i2 >> 9) & 7;
    int lane = (i2 >> 4) & 31;
    int e = i2 & 15;
    int k = 32 * c + ((e & 8) << 1) + ((lane >> 4) << 3) + (e & 7);
    int col = 16 * n + (lane & 15);
    wpack[WOUT_PACK_OFF + i2] = (_Float16)Wout[(l * 64 + k) * 128 + col];
  } else if (idx < PACK_TOTAL) {
    int i3 = idx - WS_PACK_SZ - WOUT_PACK_SZ;
    int l = i3 >> 11;                // / (4*32*16)
    int n = (i3 >> 9) & 3;
    int lane = (i3 >> 4) & 31;
    int e = i3 & 15;
    int k = ((lane >> 4) << 4) + e;  // kchunk == 0, K padded to 32
    int col = 16 * n + (lane & 15);
    float v = (k < FEAT) ? ffn[(l * FEAT + k) * 64 + col] : 0.0f;
    wpack[FFN_PACK_OFF + i3] = (_Float16)v;
  }
}

// ---------------------------------------------------------------------------
// Main fused kernel: hash-grid gather + Fourier WMMA + SIREN WMMA stack.
// 256 threads = 8 waves; each wave owns 16 points start-to-finish.
// ---------------------------------------------------------------------------
__global__ __launch_bounds__(256) void ffb_main(
    const float* __restrict__ in_pos, const float* __restrict__ table,
    const float* __restrict__ W0, const float* __restrict__ b0v,
    const float* __restrict__ bsv, const float* __restrict__ boutv,
    const _Float16* __restrict__ wpack, float* __restrict__ out) {
  __shared__ __align__(16) _Float16 lds_x[BLK_PTS * XSTR];        // 18 KB
  __shared__ __align__(16) _Float16 gridf[LVLS][BLK_PTS][FEAT];   // 8 KB

  const int tid = threadIdx.x;
  const int wave = tid >> 5;
  const int lane = tid & 31;
  const int row16 = lane & 15;
  const int half = lane >> 4;
  const int pbase = blockIdx.x * BLK_PTS + wave * 16;
  const int pt = pbase + row16;

  // BOUND == 1: SIREN input is raw pos; grid input is (pos+1)/2
  const float px = in_pos[3 * pt + 0];
  const float py = in_pos[3 * pt + 1];
  const float pz = in_pos[3 * pt + 2];
  const float qx = (px + 1.0f) * 0.5f;
  const float qy = (py + 1.0f) * 0.5f;
  const float qz = (pz + 1.0f) * 0.5f;

  // ---- hash-grid stage: lane pair (L, L+16) splits the 8 corners ----
  for (int l = 0; l < LVLS; ++l) {
    const float res = (float)(16 << l);
    float gx = qx * res, gy = qy * res, gz = qz * res;
    float x0 = floorf(gx), y0 = floorf(gy), z0 = floorf(gz);
    float fx = gx - x0, fy = gy - y0, fz = gz - z0;
    unsigned ux = (unsigned)x0, uy = (unsigned)y0, uz = (unsigned)z0;
    float acc[FEAT];
#pragma unroll
    for (int k = 0; k < FEAT; ++k) acc[k] = 0.0f;
    const int cb = half * 4;
#pragma unroll
    for (int ci = 0; ci < 4; ++ci) {
      int c = cb + ci;
      unsigned cx = (c >> 2) & 1, cy = (c >> 1) & 1, cz = c & 1;
      unsigned h =
          ((ux + cx) * 1u) ^ ((uy + cy) * 2654435761u) ^ ((uz + cz) * 805459861u);
      h &= TMASK;
      const float4* fp =
          (const float4*)(table + (((size_t)l << 19) + h) * FEAT);
      float4 f0 = fp[0], f1 = fp[1];
      float w = (cx ? fx : 1.0f - fx) * (cy ? fy : 1.0f - fy) *
                (cz ? fz : 1.0f - fz);
      acc[0] += w * f0.x; acc[1] += w * f0.y; acc[2] += w * f0.z; acc[3] += w * f0.w;
      acc[4] += w * f1.x; acc[5] += w * f1.y; acc[6] += w * f1.z; acc[7] += w * f1.w;
    }
#pragma unroll
    for (int k = 0; k < FEAT; ++k) acc[k] += __shfl_xor(acc[k], 16, 32);
    if (half == 0) {
      union { _Float16 h8[8]; float4 f4; } u;
#pragma unroll
      for (int k = 0; k < FEAT; ++k) u.h8[k] = (_Float16)acc[k];
      *(float4*)&gridf[l][wave * 16 + row16][0] = u.f4;
    }
  }

  // ---- SIREN layer 0 (K=3, plain VALU) -> lds_x as f16 ----
#pragma unroll 4
  for (int it = 0; it < 32; ++it) {
    int j = half * 32 + it;
    float t = px * W0[j] + py * W0[64 + j] + pz * W0[128 + j] + b0v[j];
    lds_x[(wave * 16 + row16) * XSTR + j] = (_Float16)__sinf(30.0f * t);
  }
  __syncthreads();

  // A-fragment loads: 16-bit A 16x32 layout -> lane row = lane&15,
  // elements 0..7 <-> K = kbase + 8*(lane>>4) + e, 8..15 <-> +16.
  const int khalf = half * 8;
  const _Float16* xrow = &lds_x[(wave * 16 + row16) * XSTR];
  v16h a0, a1;
  {
    union { v16h h; float4 f4[2]; } u;
    u.f4[0] = *(const float4*)(xrow + khalf);
    u.f4[1] = *(const float4*)(xrow + khalf + 16);
    a0 = u.h;
    u.f4[0] = *(const float4*)(xrow + 32 + khalf);
    u.f4[1] = *(const float4*)(xrow + 32 + khalf + 16);
    a1 = u.h;
  }

  float xo[8][8];
#pragma unroll
  for (int n = 0; n < 8; ++n)
#pragma unroll
    for (int r = 0; r < 8; ++r) xo[n][r] = 0.0f;

  for (int l = 0; l < LVLS; ++l) {
    // Fourier branch A fragment (K padded to 32 with zeros)
    v16h ga;
#pragma unroll
    for (int e = 0; e < 16; ++e) ga[e] = (_Float16)0.0f;
    if (half == 0) {
      union { v16h h; float4 f4[2]; } u;
      u.h = ga;
      u.f4[0] = *(const float4*)&gridf[l][wave * 16 + row16][0];
      ga = u.h;
    }
    v8f embt[4];
#pragma unroll
    for (int n = 0; n < 4; ++n) {
      union { v16h h; float4 f4[2]; } bu;
      const float4* bp =
          (const float4*)(wpack + FFN_PACK_OFF + (((l * 4 + n) * 32 + lane) << 4));
      bu.f4[0] = bp[0]; bu.f4[1] = bp[1];
      v8f c = {};
      c = __builtin_amdgcn_wmma_f32_16x16x32_f16(false, ga, false, bu.h,
                                                 (short)0, c, false, false);
      embt[n] = c;
    }

    // x = sin(30*(x @ Ws + bs)) + sin(2pi * emb)
#pragma unroll
    for (int n = 0; n < 4; ++n) {
      union { v16h h; float4 f4[2]; } bu;
      const float4* bp0 =
          (const float4*)(wpack + ((((l * 2 + 0) * 4 + n) * 32 + lane) << 4));
      bu.f4[0] = bp0[0]; bu.f4[1] = bp0[1];
      v8f c = {};
      c = __builtin_amdgcn_wmma_f32_16x16x32_f16(false, a0, false, bu.h,
                                                 (short)0, c, false, false);
      const float4* bp1 =
          (const float4*)(wpack + ((((l * 2 + 1) * 4 + n) * 32 + lane) << 4));
      bu.f4[0] = bp1[0]; bu.f4[1] = bp1[1];
      c = __builtin_amdgcn_wmma_f32_16x16x32_f16(false, a1, false, bu.h,
                                                 (short)0, c, false, false);
      int col = n * 16 + row16;
      float bb = bsv[l * 64 + col];
#pragma unroll
      for (int r = 0; r < 8; ++r) {
        float v = __sinf(30.0f * (c[r] + bb)) +
                  __sinf(6.28318530718f * embt[n][r]);
        lds_x[(wave * 16 + half * 8 + r) * XSTR + col] = (_Float16)v;
      }
    }
    __syncthreads();

    // reload A fragments of updated x (also reused by next level's Ws GEMM)
    {
      union { v16h h; float4 f4[2]; } u;
      u.f4[0] = *(const float4*)(xrow + khalf);
      u.f4[1] = *(const float4*)(xrow + khalf + 16);
      a0 = u.h;
      u.f4[0] = *(const float4*)(xrow + 32 + khalf);
      u.f4[1] = *(const float4*)(xrow + 32 + khalf + 16);
      a1 = u.h;
    }

    // x_out += sin(30*(x @ Wout + bout))
#pragma unroll
    for (int n = 0; n < 8; ++n) {
      union { v16h h; float4 f4[2]; } bu;
      const float4* bp0 = (const float4*)(
          wpack + WOUT_PACK_OFF + ((((l * 2 + 0) * 8 + n) * 32 + lane) << 4));
      bu.f4[0] = bp0[0]; bu.f4[1] = bp0[1];
      v8f c = {};
      c = __builtin_amdgcn_wmma_f32_16x16x32_f16(false, a0, false, bu.h,
                                                 (short)0, c, false, false);
      const float4* bp1 = (const float4*)(
          wpack + WOUT_PACK_OFF + ((((l * 2 + 1) * 8 + n) * 32 + lane) << 4));
      bu.f4[0] = bp1[0]; bu.f4[1] = bp1[1];
      c = __builtin_amdgcn_wmma_f32_16x16x32_f16(false, a1, false, bu.h,
                                                 (short)0, c, false, false);
      int col = n * 16 + row16;
      float bb = boutv[l * 128 + col];
#pragma unroll
      for (int r = 0; r < 8; ++r) xo[n][r] += __sinf(30.0f * (c[r] + bb));
    }
  }

  // ---- store output tile (half-wave coalesced along columns) ----
#pragma unroll
  for (int n = 0; n < 8; ++n) {
    int col = n * 16 + row16;
#pragma unroll
    for (int r = 0; r < 8; ++r) {
      int prow = pbase + half * 8 + r;
      out[(size_t)prow * ODIM + col] = xo[n][r];
    }
  }
}

extern "C" void kernel_launch(void* const* d_in, const int* in_sizes, int n_in,
                              void* d_out, int out_size, void* d_ws,
                              size_t ws_size, hipStream_t stream) {
  const float* in_pos = (const float*)d_in[0];
  const float* table = (const float*)d_in[1];
  const float* ffn = (const float*)d_in[2];
  const float* W0 = (const float*)d_in[3];
  const float* b0 = (const float*)d_in[4];
  const float* Ws = (const float*)d_in[5];
  const float* bs = (const float*)d_in[6];
  const float* Wout = (const float*)d_in[7];
  const float* bout = (const float*)d_in[8];
  _Float16* wpack = (_Float16*)d_ws;
  float* out = (float*)d_out;

  ffb_prep<<<(PACK_TOTAL + 255) / 256, 256, 0, stream>>>(ffn, Ws, Wout, wpack);
  ffb_main<<<NPTS / BLK_PTS, 256, 0, stream>>>(in_pos, table, W0, b0, bs, bout,
                                               wpack, out);
}
